// Block_17712445129207
// MI455X (gfx1250) — matrix operand order repaired and checked by reading, not compiled, not run on hardware
//
#include <hip/hip_runtime.h>

typedef __attribute__((ext_vector_type(16))) _Float16 v16h;
typedef __attribute__((ext_vector_type(8)))  float    v8f;

#define HEADS   4
#define EMBED   400
#define CP      416      // EMBED padded to mult of 32 (K dim)
#define HD      100
#define DP      128      // head dim padded to mult of 32
#define SEQ     2048
#define BATCH   8
#define MTOK    (BATCH*SEQ)   // 16384
#define DFF     1600

// ---------------- fragment loaders (f16 WMMA 16x16x32) ----------------
// A-matrix 16x32 (MxK): lane l: M=l%16; halves i=0..7 -> K=g*8+i, i=8..15 -> K=16+g*8+(i-8)
static __device__ __forceinline__ v16h load_frag_a(const _Float16* base, int ld, int k0) {
  const int l = threadIdx.x & 31;
  const _Float16* p = base + (size_t)(l & 15) * ld + k0 + ((l >> 4) << 3);
  union { v16h v; uint4 q[2]; } u;
  u.q[0] = *(const uint4*)(p);
  u.q[1] = *(const uint4*)(p + 16);
  return u.v;
}
// B-matrix 32x16 (KxN), loaded from BT[N][K] row-major: lane l: N=l%16; i=0..7 -> K=g*8+i, i=8..15 -> K=16+g*8+(i-8)
static __device__ __forceinline__ v16h load_frag_b(const _Float16* base, int ld, int k0) {
  const int l = threadIdx.x & 31;
  const _Float16* p = base + (size_t)(l & 15) * ld + k0 + ((l >> 4) << 3);
  union { v16h v; uint4 q[2]; } u;
  u.q[0] = *(const uint4*)(p);
  u.q[1] = *(const uint4*)(p + 16);
  return u.v;
}
static __device__ __forceinline__ v8f wmma_f16(v16h a, v16h b, v8f c) {
  return __builtin_amdgcn_wmma_f32_16x16x32_f16(false, a, false, b, (short)0, c, false, false);
}

// ---------------- weight transpose + pad to f16 ----------------
// src [K][N] f32 row-major -> dst [Np][Kp] f16, dst[n][k] = (k<K && n<N) ? src[k][n] : 0
__global__ void transpose_pad(const float* __restrict__ src, _Float16* __restrict__ dst,
                              int K, int N, int Kp, int Np) {
  int zz = blockIdx.z;
  int idx = blockIdx.x * 256 + threadIdx.x;
  if (idx >= Np * Kp) return;
  int n = idx / Kp, k = idx % Kp;
  float v = (n < N && k < K) ? src[(size_t)zz * K * N + (size_t)k * N + n] : 0.0f;
  dst[(size_t)zz * Np * Kp + idx] = (_Float16)v;
}

// ---------------- LayerNorm: one wave per token, f32 in, f16 (padded) out ----------------
__global__ void __launch_bounds__(256)
ln_kernel(const float* __restrict__ X, const float* __restrict__ gamma,
          const float* __restrict__ beta, _Float16* __restrict__ H16) {
  int row  = blockIdx.x * 8 + (threadIdx.x >> 5);
  int lane = threadIdx.x & 31;
  const float* xr = X + (size_t)row * EMBED;
  float s = 0.f, ss = 0.f;
  for (int c = lane; c < EMBED; c += 32) { float v = xr[c]; s += v; ss += v * v; }
#pragma unroll
  for (int off = 16; off; off >>= 1) { s += __shfl_xor(s, off); ss += __shfl_xor(ss, off); }
  float mu  = s * (1.0f / EMBED);
  float var = ss * (1.0f / EMBED) - mu * mu;
  float rs  = rsqrtf(var + 1e-5f);
  _Float16* hr = H16 + (size_t)row * CP;
  for (int c = lane; c < EMBED; c += 32)
    hr[c] = (_Float16)((xr[c] - mu) * rs * gamma[c] + beta[c]);
  if (lane < CP - EMBED) hr[EMBED + lane] = (_Float16)0.0f;
}

// ---------------- QKV projection GEMM (register-blocked 64x64 per wave) ----------------
// grid.z = mat*HEADS + h (mat: 0=Q,1=K,2=V). Each wave: 4 m-tiles x 4 n-tiles. K=CP, N=DP.
__global__ void __launch_bounds__(128)
qkv_gemm(const _Float16* __restrict__ h16,
         const _Float16* __restrict__ wqT, const _Float16* __restrict__ wkT,
         const _Float16* __restrict__ wvT,
         _Float16* __restrict__ q16, _Float16* __restrict__ k16, _Float16* __restrict__ vT) {
  const int NG = 2;                             // 128 / 64
  int wave = threadIdx.x >> 5;
  int job  = blockIdx.x * 4 + wave;             // 512 jobs per z
  int mg = job / NG, ng = job % NG;
  int z = blockIdx.z, mat = z >> 2, h = z & 3;
  const _Float16* WT = (mat == 0 ? wqT : (mat == 1 ? wkT : wvT)) + (size_t)h * DP * CP;
  int m0 = mg * 64, n0 = ng * 64;
  v8f acc[4][4] = {};
  for (int k0 = 0; k0 < CP; k0 += 32) {
    v16h a[4];
#pragma unroll
    for (int mt = 0; mt < 4; ++mt)
      a[mt] = load_frag_a(h16 + (size_t)(m0 + mt * 16) * CP, CP, k0);
#pragma unroll
    for (int t = 0; t < 4; ++t) {
      v16h bb = load_frag_b(WT + (size_t)(n0 + t * 16) * CP, CP, k0);
#pragma unroll
      for (int mt = 0; mt < 4; ++mt) acc[mt][t] = wmma_f16(a[mt], bb, acc[mt][t]);
    }
  }
  int l = threadIdx.x & 31, lo = l & 15, g = l >> 4;
  int b_  = m0 >> 11;                 // 64-row groups never straddle a batch (2048%64==0)
  int tt0 = m0 & 2047;
  size_t bh = (size_t)(b_ * HEADS + h);
  if (mat < 2) {
    _Float16* outp = (mat == 0 ? q16 : k16) + bh * SEQ * DP;
#pragma unroll
    for (int mt = 0; mt < 4; ++mt)
#pragma unroll
      for (int t = 0; t < 4; ++t) {
        int n = n0 + t * 16 + lo;
#pragma unroll
        for (int r = 0; r < 8; ++r)
          outp[(size_t)(tt0 + mt * 16 + r + 8 * g) * DP + n] = (_Float16)acc[mt][t][r];
      }
  } else {                             // V stored transposed: vT[b,h][d][t]
    _Float16* outp = vT + bh * DP * SEQ;
#pragma unroll
    for (int mt = 0; mt < 4; ++mt)
#pragma unroll
      for (int t = 0; t < 4; ++t) {
        int n = n0 + t * 16 + lo;
        _Float16* col = outp + (size_t)n * SEQ + tt0 + mt * 16 + 8 * g;
#pragma unroll
        for (int r = 0; r < 8; ++r) col[r] = (_Float16)acc[mt][t][r];
      }
  }
}

// ---------------- fused causal attention (flash-style, 1 wave / 32 query rows) ----------------
__global__ void __launch_bounds__(32)
attention_kernel(const _Float16* __restrict__ q16, const _Float16* __restrict__ k16,
                 const _Float16* __restrict__ vT, _Float16* __restrict__ att16) {
  __shared__ __align__(16) _Float16 plds[2 * 16 * 32];
  int bh = blockIdx.y;                 // b*HEADS + h
  int h  = bh & 3;
  int q0 = blockIdx.x * 32;
  int l = threadIdx.x, lo = l & 15, g = l >> 4;
  const _Float16* Q  = q16 + (size_t)bh * SEQ * DP + (size_t)q0 * DP;
  const _Float16* Kb = k16 + (size_t)bh * SEQ * DP;
  const _Float16* V  = vT  + (size_t)bh * DP * SEQ;

  v16h qf[2][4];
#pragma unroll
  for (int m = 0; m < 2; ++m)
#pragma unroll
    for (int kk = 0; kk < 4; ++kk)
      qf[m][kk] = load_frag_a(Q + (size_t)(m * 16) * DP, DP, kk * 32);

  v8f acc[2][8] = {};
  float m_i[2][8], l_i[2][8];
#pragma unroll
  for (int m = 0; m < 2; ++m)
#pragma unroll
    for (int r = 0; r < 8; ++r) { m_i[m][r] = -1e30f; l_i[m][r] = 0.0f; }
  const float sc = 0.1f;               // 1/sqrt(100)

  for (int s0 = 0; s0 <= q0 + 31; s0 += 32) {
    v8f S[2][2] = {};
#pragma unroll
    for (int kk = 0; kk < 4; ++kk) {
      v16h b0 = load_frag_b(Kb + (size_t)s0 * DP, DP, kk * 32);
      v16h b1 = load_frag_b(Kb + (size_t)(s0 + 16) * DP, DP, kk * 32);
#pragma unroll
      for (int m = 0; m < 2; ++m) {
        S[m][0] = wmma_f16(qf[m][kk], b0, S[m][0]);
        S[m][1] = wmma_f16(qf[m][kk], b1, S[m][1]);
      }
    }
#pragma unroll
    for (int m = 0; m < 2; ++m) {
#pragma unroll
      for (int r = 0; r < 8; ++r) {
        int qrow = q0 + m * 16 + r + 8 * g;
        float v0 = ((s0 + lo)      <= qrow) ? S[m][0][r] * sc : -1e30f;
        float v1 = ((s0 + 16 + lo) <= qrow) ? S[m][1][r] * sc : -1e30f;
        float mx = fmaxf(v0, v1);
#pragma unroll
        for (int off = 8; off; off >>= 1) mx = fmaxf(mx, __shfl_xor(mx, off));
        float mnew  = fmaxf(m_i[m][r], mx);
        float alpha = __expf(m_i[m][r] - mnew);
        float p0 = __expf(v0 - mnew);
        float p1 = __expf(v1 - mnew);
        float rs = p0 + p1;
#pragma unroll
        for (int off = 8; off; off >>= 1) rs += __shfl_xor(rs, off);
        l_i[m][r] = l_i[m][r] * alpha + rs;
        m_i[m][r] = mnew;
#pragma unroll
        for (int n = 0; n < 8; ++n) acc[m][n][r] *= alpha;
        plds[m * 512 + (r + 8 * g) * 32 + lo]      = (_Float16)p0;
        plds[m * 512 + (r + 8 * g) * 32 + 16 + lo] = (_Float16)p1;
      }
    }
    __syncthreads();                           // stores visible (single-wave block)
    v16h pf0 = load_frag_a(plds,       32, 0);
    v16h pf1 = load_frag_a(plds + 512, 32, 0);
    __syncthreads();                           // reads done before next overwrite
#pragma unroll
    for (int n = 0; n < 8; ++n) {
      v16h vf = load_frag_b(V + (size_t)(n * 16) * SEQ, SEQ, s0);
      acc[0][n] = wmma_f16(pf0, vf, acc[0][n]);
      acc[1][n] = wmma_f16(pf1, vf, acc[1][n]);
    }
  }
  int b_ = bh >> 2;
#pragma unroll
  for (int m = 0; m < 2; ++m)
#pragma unroll
    for (int n = 0; n < 7; ++n) {
      int d = n * 16 + lo;
      if (d < HD) {
#pragma unroll
        for (int r = 0; r < 8; ++r) {
          int qrow = q0 + m * 16 + r + 8 * g;
          att16[(size_t)(b_ * SEQ + qrow) * CP + h * HD + d] =
              (_Float16)(acc[m][n][r] / l_i[m][r]);
        }
      }
    }
}

// ---------------- output projection + residual (64x80 per wave) ----------------
__global__ void __launch_bounds__(128)
gemm_wo(const _Float16* __restrict__ att16, const _Float16* __restrict__ woT,
        const float* __restrict__ x, const float* __restrict__ bo, float* __restrict__ x2) {
  const int NG = 5;                      // 400 / 80
  int wave = threadIdx.x >> 5;
  int job  = blockIdx.x * 4 + wave;      // 1280 jobs
  int mg = job / NG, ng = job % NG;
  int m0 = mg * 64, n0 = ng * 80;
  v8f acc[4][5] = {};
  for (int k0 = 0; k0 < CP; k0 += 32) {
    v16h a[4];
#pragma unroll
    for (int mt = 0; mt < 4; ++mt)
      a[mt] = load_frag_a(att16 + (size_t)(m0 + mt * 16) * CP, CP, k0);
#pragma unroll
    for (int t = 0; t < 5; ++t) {
      v16h bb = load_frag_b(woT + (size_t)(n0 + t * 16) * CP, CP, k0);
#pragma unroll
      for (int mt = 0; mt < 4; ++mt) acc[mt][t] = wmma_f16(a[mt], bb, acc[mt][t]);
    }
  }
  int l = threadIdx.x & 31, lo = l & 15, g = l >> 4;
#pragma unroll
  for (int mt = 0; mt < 4; ++mt)
#pragma unroll
    for (int t = 0; t < 5; ++t) {
      int n = n0 + t * 16 + lo;
#pragma unroll
      for (int r = 0; r < 8; ++r) {
        size_t row = (size_t)m0 + mt * 16 + r + 8 * g;
        x2[row * EMBED + n] = x[row * EMBED + n] + bo[n] + acc[mt][t][r];
      }
    }
}

// ---------------- FFN layer 1: relu(h2 @ W1 + b1) -> f16 (64x80 per wave) ----------------
__global__ void __launch_bounds__(128)
gemm_w1(const _Float16* __restrict__ h2, const _Float16* __restrict__ w1T,
        const float* __restrict__ b1, _Float16* __restrict__ ff1) {
  const int NG = 20;                     // 1600 / 80
  int wave = threadIdx.x >> 5;
  int job  = blockIdx.x * 4 + wave;      // 5120 jobs
  int mg = job / NG, ng = job % NG;
  int m0 = mg * 64, n0 = ng * 80;
  v8f acc[4][5] = {};
  for (int k0 = 0; k0 < CP; k0 += 32) {
    v16h a[4];
#pragma unroll
    for (int mt = 0; mt < 4; ++mt)
      a[mt] = load_frag_a(h2 + (size_t)(m0 + mt * 16) * CP, CP, k0);
#pragma unroll
    for (int t = 0; t < 5; ++t) {
      v16h bb = load_frag_b(w1T + (size_t)(n0 + t * 16) * CP, CP, k0);
#pragma unroll
      for (int mt = 0; mt < 4; ++mt) acc[mt][t] = wmma_f16(a[mt], bb, acc[mt][t]);
    }
  }
  int l = threadIdx.x & 31, lo = l & 15, g = l >> 4;
#pragma unroll
  for (int mt = 0; mt < 4; ++mt)
#pragma unroll
    for (int t = 0; t < 5; ++t) {
      int n = n0 + t * 16 + lo;
#pragma unroll
      for (int r = 0; r < 8; ++r) {
        size_t row = (size_t)m0 + mt * 16 + r + 8 * g;
        ff1[row * DFF + n] = (_Float16)fmaxf(acc[mt][t][r] + b1[n], 0.0f);
      }
    }
}

// ---------------- FFN layer 2 + final residual -> d_out (64x80 per wave) ----------------
__global__ void __launch_bounds__(128)
gemm_w2(const _Float16* __restrict__ ff1, const _Float16* __restrict__ w2T,
        const float* __restrict__ b2, const float* __restrict__ x2, float* __restrict__ outp) {
  const int NG = 5;                      // 400 / 80
  int wave = threadIdx.x >> 5;
  int job  = blockIdx.x * 4 + wave;      // 1280 jobs
  int mg = job / NG, ng = job % NG;
  int m0 = mg * 64, n0 = ng * 80;
  v8f acc[4][5] = {};
  for (int k0 = 0; k0 < DFF; k0 += 32) {
    v16h a[4];
#pragma unroll
    for (int mt = 0; mt < 4; ++mt)
      a[mt] = load_frag_a(ff1 + (size_t)(m0 + mt * 16) * DFF, DFF, k0);
#pragma unroll
    for (int t = 0; t < 5; ++t) {
      v16h bb = load_frag_b(w2T + (size_t)(n0 + t * 16) * DFF, DFF, k0);
#pragma unroll
      for (int mt = 0; mt < 4; ++mt) acc[mt][t] = wmma_f16(a[mt], bb, acc[mt][t]);
    }
  }
  int l = threadIdx.x & 31, lo = l & 15, g = l >> 4;
#pragma unroll
  for (int mt = 0; mt < 4; ++mt)
#pragma unroll
    for (int t = 0; t < 5; ++t) {
      int n = n0 + t * 16 + lo;
#pragma unroll
      for (int r = 0; r < 8; ++r) {
        size_t row = (size_t)m0 + mt * 16 + r + 8 * g;
        outp[row * EMBED + n] = x2[row * EMBED + n] + b2[n] + acc[mt][t][r];
      }
    }
}

// ---------------- host orchestration ----------------
extern "C" void kernel_launch(void* const* d_in, const int* in_sizes, int n_in,
                              void* d_out, int out_size, void* d_ws, size_t ws_size,
                              hipStream_t stream) {
  (void)in_sizes; (void)n_in; (void)out_size; (void)ws_size;
  const float* x    = (const float*)d_in[0];
  const float* ln1g = (const float*)d_in[1];
  const float* ln1b = (const float*)d_in[2];
  const float* ln2g = (const float*)d_in[3];
  const float* ln2b = (const float*)d_in[4];
  const float* Wq   = (const float*)d_in[5];
  const float* Wk   = (const float*)d_in[6];
  const float* Wv   = (const float*)d_in[7];
  const float* Wo   = (const float*)d_in[8];
  const float* bo   = (const float*)d_in[9];
  const float* W1   = (const float*)d_in[10];
  const float* b1   = (const float*)d_in[11];
  const float* W2   = (const float*)d_in[12];
  const float* b2   = (const float*)d_in[13];
  float* outp = (float*)d_out;

  char* ws = (char*)d_ws;
  size_t off = 0;
  auto take = [&](size_t bytes) -> char* {
    char* p = ws + off; off += (bytes + 255) & ~(size_t)255; return p;
  };
  _Float16* h16   = (_Float16*)take((size_t)MTOK * CP * 2);     // LN1 out; reused as LN2 out
  char*     qkvR  = take(52428800);                             // q,k,vT; reused as ff1
  _Float16* q16   = (_Float16*)(qkvR);
  _Float16* k16   = (_Float16*)(qkvR + 16777216);
  _Float16* vT    = (_Float16*)(qkvR + 33554432);
  _Float16* ff1   = (_Float16*)(qkvR);
  _Float16* att16 = (_Float16*)take((size_t)MTOK * CP * 2);
  float*    x2    = (float*)take((size_t)MTOK * EMBED * 4);
  _Float16* wqT   = (_Float16*)take((size_t)HEADS * DP * CP * 2);
  _Float16* wkT   = (_Float16*)take((size_t)HEADS * DP * CP * 2);
  _Float16* wvT   = (_Float16*)take((size_t)HEADS * DP * CP * 2);
  _Float16* woT   = (_Float16*)take((size_t)EMBED * CP * 2);
  _Float16* w1T   = (_Float16*)take((size_t)DFF * CP * 2);
  _Float16* w2T   = (_Float16*)take((size_t)EMBED * DFF * 2);

  // Weight conversion (transpose + pad + f32->f16)
  transpose_pad<<<dim3((DP * CP + 255) / 256, 1, HEADS), 256, 0, stream>>>(Wq, wqT, EMBED, HD, CP, DP);
  transpose_pad<<<dim3((DP * CP + 255) / 256, 1, HEADS), 256, 0, stream>>>(Wk, wkT, EMBED, HD, CP, DP);
  transpose_pad<<<dim3((DP * CP + 255) / 256, 1, HEADS), 256, 0, stream>>>(Wv, wvT, EMBED, HD, CP, DP);
  transpose_pad<<<dim3((EMBED * CP + 255) / 256, 1, 1), 256, 0, stream>>>(Wo, woT, EMBED, EMBED, CP, EMBED);
  transpose_pad<<<dim3((DFF * CP + 255) / 256, 1, 1), 256, 0, stream>>>(W1, w1T, EMBED, DFF, CP, DFF);
  transpose_pad<<<dim3((EMBED * DFF + 255) / 256, 1, 1), 256, 0, stream>>>(W2, w2T, DFF, EMBED, DFF, EMBED);

  // LN1
  ln_kernel<<<MTOK / 8, 256, 0, stream>>>(x, ln1g, ln1b, h16);
  // QKV projections (grid.z = mat*HEADS + h); 512 wave-jobs per z / 4 waves per block
  qkv_gemm<<<dim3(128, 1, 12), 128, 0, stream>>>(h16, wqT, wkT, wvT, q16, k16, vT);
  // zero att16 (incl. pad cols 400..415) before attention writes cols 0..399
  hipMemsetAsync(att16, 0, (size_t)MTOK * CP * 2, stream);
  // fused causal attention (32 query rows per wave)
  attention_kernel<<<dim3(SEQ / 32, BATCH * HEADS), 32, 0, stream>>>(q16, k16, vT, att16);
  // output projection + residual -> x2   (1280 jobs / 4)
  gemm_wo<<<320, 128, 0, stream>>>(att16, woT, x, bo, x2);
  // LN2 (reuse h16)
  ln_kernel<<<MTOK / 8, 256, 0, stream>>>(x2, ln2g, ln2b, h16);
  // FFN
  gemm_w1<<<1280, 128, 0, stream>>>(h16, w1T, b1, ff1);   // 5120 jobs / 4
  gemm_w2<<<320, 128, 0, stream>>>(ff1, w2T, b2, x2, outp); // 1280 jobs / 4
}
